// MVURE_Layer_28836410425902
// MI455X (gfx1250) — compile-verified, matrix operand order checked by reading, not run on hardware
//
#include <hip/hip_runtime.h>

#define NN   50000
#define DD   128
#define OUTD 128
#define HH   8
#define FF   16
#define ALPHA_C 0.8f
#define BETA_C  0.5f

typedef __attribute__((ext_vector_type(16))) _Float16 v16h;
typedef __attribute__((ext_vector_type(8)))  _Float16 v8h;
typedef __attribute__((ext_vector_type(8)))  float    v8f;

// ---------------- workspace layout (bytes, all 256B aligned) ----------------
#define OFF_XB    0ull                     // N*D f16            = 12,800,000
#define OFF_WB    12800000ull              // 3*OUT*D f16        =     98,304
#define OFF_VIEWS 12898304ull              // 3*N*OUT f32        = 76,800,000
#define OFF_FEAT  89698304ull              // N*OUT f32          = 25,600,000
#define OFF_EL    115298304ull             // N*H f32
#define OFF_ER    116898304ull             // N*H f32
#define OFF_M     118498304ull             // N*H u32 (enc max)
#define OFF_S     120098304ull             // N*H f32
#define OFF_SCAL  121698304ull             // 12 f32: [0..2]=logit acc, [3..5]=w, [6..8]=omega acc, [9..11]=omega

// LDS pitch for staged W: 128 + 8 halfs -> 272B rows (68 dwords), so the 16
// rows a wave touches per ds_load_b128 land on disjoint bank groups.
#define WPITCH 136

// order-preserving float<->uint encoding (memset-0 == -inf identity)
__device__ __forceinline__ unsigned fenc(float f) {
  unsigned u = __float_as_uint(f);
  return (u & 0x80000000u) ? ~u : (u | 0x80000000u);
}
__device__ __forceinline__ float fdec(unsigned u) {
  return (u & 0x80000000u) ? __uint_as_float(u & 0x7FFFFFFFu) : __uint_as_float(~u);
}

// ---------------- conversion kernels ----------------
__global__ __launch_bounds__(256) void conv_x_kernel(const float* __restrict__ x,
                                                     _Float16* __restrict__ xb) {
  int i = blockIdx.x * 256 + threadIdx.x;
  if (i < NN * DD) xb[i] = (_Float16)x[i];
}

__global__ __launch_bounds__(256) void conv_w_kernel(const float* __restrict__ w0,
                                                     const float* __restrict__ w1,
                                                     const float* __restrict__ w2,
                                                     _Float16* __restrict__ wb) {
  int i = blockIdx.x * 256 + threadIdx.x;
  if (i < 3 * OUTD * DD) {
    int v = i >> 14, j = i & 16383;  // OUT*D = 16384
    const float* p = (v == 0) ? w0 : ((v == 1) ? w1 : w2);
    wb[i] = (_Float16)p[j];
  }
}

// ---------------- WMMA GEMM: feat[N,OUT] = x[N,D] @ W[OUT,D]^T ----------------
// 5 waves/block (3125 row tiles = 625 blocks x 5, exact -> no guards, EXEC all-1s).
// W (32KB f16) staged once per block into padded LDS; each wave computes a full
// 16x128 stripe: 8 accumulators, A fragment loaded once per K step, 32 WMMAs.
__global__ __launch_bounds__(160)
void gat_feat_wmma(const _Float16* __restrict__ xb,
                   const _Float16* __restrict__ wb,   // this view's W, [OUT,D] row-major f16
                   float* __restrict__ feat) {
  __shared__ alignas(16) _Float16 lds_w[OUTD * WPITCH];

  // cooperative stage: 128 rows x 16 chunks of 8 halfs = 2048 chunks
  const int tid = threadIdx.y * 32 + threadIdx.x;
  for (int c = tid; c < OUTD * 16; c += 160) {
    int row = c >> 4, c8 = (c & 15) * 8;
    *(v8h*)(&lds_w[row * WPITCH + c8]) = *(const v8h*)(wb + row * DD + c8);
  }
  __syncthreads();

  const int lane  = threadIdx.x;        // 0..31
  const int half  = lane >> 4;          // 0/1
  const int idx16 = lane & 15;
  const int mt = blockIdx.x * 5 + threadIdx.y;    // row tile, 0..3124

  const _Float16* arow = xb + (size_t)(mt * 16 + idx16) * DD; // A row (node)

  v8f acc[8] = {};
#pragma unroll
  for (int kt = 0; kt < DD / 32; ++kt) {
    const int k0 = kt * 32 + half * 8;
    v8h a0 = *(const v8h*)(arow + k0);
    v8h a1 = *(const v8h*)(arow + k0 + 16);
    v16h A;
#pragma unroll
    for (int i = 0; i < 8; ++i) { A[i] = a0[i]; A[8 + i] = a1[i]; }
#pragma unroll
    for (int ot = 0; ot < 8; ++ot) {
      const _Float16* bp = &lds_w[(ot * 16 + idx16) * WPITCH + k0];
      v8h b0 = *(const v8h*)bp;
      v8h b1 = *(const v8h*)(bp + 16);
      v16h B;
#pragma unroll
      for (int i = 0; i < 8; ++i) { B[i] = b0[i]; B[8 + i] = b1[i]; }
      acc[ot] = __builtin_amdgcn_wmma_f32_16x16x32_f16(false, A, false, B, (short)0,
                                                       acc[ot], false, false);
    }
  }
  // D layout: VGPR r -> row M = half*8 + r, col N = idx16
  float* outb = feat + (size_t)(mt * 16 + half * 8) * OUTD + idx16;
#pragma unroll
  for (int ot = 0; ot < 8; ++ot)
#pragma unroll
    for (int r = 0; r < 8; ++r) outb[(size_t)r * OUTD + ot * 16] = acc[ot][r];
}

// ---------------- attention logits el/er per (node, head) ----------------
__global__ __launch_bounds__(256)
void elr_kernel(const float* __restrict__ feat, const float* __restrict__ al,
                const float* __restrict__ ar, float* __restrict__ el,
                float* __restrict__ er) {
  int i = blockIdx.x * 256 + threadIdx.x;
  if (i >= NN * HH) return;
  int n = i >> 3, h = i & 7;
  const float* fp = feat + (size_t)n * OUTD + h * FF;
  const float* alp = al + h * FF;
  const float* arp = ar + h * FF;
  float accl = 0.f, accr = 0.f;
#pragma unroll
  for (int f = 0; f < FF; ++f) { float v = fp[f]; accl = fmaf(v, alp[f], accl); accr = fmaf(v, arp[f], accr); }
  el[i] = accl; er[i] = accr;
}

// ---------------- edge pass 1: segment max ----------------
__global__ __launch_bounds__(256)
void edge_max_kernel(const int* __restrict__ src, const int* __restrict__ dst,
                     const float* __restrict__ el, const float* __restrict__ er,
                     unsigned* __restrict__ menc, int E) {
  int i = blockIdx.x * 256 + threadIdx.x;
  if (i >= E * HH) return;
  int e = i >> 3, h = i & 7;
  int sN = src[e], dN = dst[e];
  float ev = el[sN * HH + h] + er[dN * HH + h];
  ev = ev > 0.f ? ev : 0.2f * ev;            // leaky_relu(0.2)
  atomicMax(&menc[dN * HH + h], fenc(ev));
}

// ---------------- edge pass 2: segment sum of exp ----------------
__global__ __launch_bounds__(256)
void edge_sum_kernel(const int* __restrict__ src, const int* __restrict__ dst,
                     const float* __restrict__ el, const float* __restrict__ er,
                     const unsigned* __restrict__ menc, float* __restrict__ s, int E) {
  int i = blockIdx.x * 256 + threadIdx.x;
  if (i >= E * HH) return;
  int e = i >> 3, h = i & 7;
  int sN = src[e], dN = dst[e];
  float ev = el[sN * HH + h] + er[dN * HH + h];
  ev = ev > 0.f ? ev : 0.2f * ev;
  float ex = __expf(ev - fdec(menc[dN * HH + h]));
  atomicAdd(&s[dN * HH + h], ex);
}

// ---------------- edge pass 3: weighted aggregate (wave per edge) ----------------
__global__ __launch_bounds__(256)
void edge_agg_kernel(const int* __restrict__ src, const int* __restrict__ dst,
                     const float* __restrict__ el, const float* __restrict__ er,
                     const unsigned* __restrict__ menc, const float* __restrict__ s,
                     const float* __restrict__ feat, float* __restrict__ aggv, int E) {
  int e = blockIdx.x * 8 + (threadIdx.x >> 5);
  if (e >= E) return;
  int j = threadIdx.x & 31;                    // lane: channels 4j..4j+3, head j/4
  int sN = src[e], dN = dst[e];
  int h = j >> 2;
  float ev = el[sN * HH + h] + er[dN * HH + h];
  ev = ev > 0.f ? ev : 0.2f * ev;
  float attn = __expf(ev - fdec(menc[dN * HH + h])) / s[dN * HH + h];
  const float4 f4 = *(const float4*)(feat + (size_t)sN * OUTD + j * 4);
  float* base = aggv + (size_t)dN * OUTD + j * 4;
  atomicAdd(base + 0, f4.x * attn);
  atomicAdd(base + 1, f4.y * attn);
  atomicAdd(base + 2, f4.z * attn);
  atomicAdd(base + 3, f4.w * attn);
}

// ---------------- bias + relu (in place on one view) ----------------
__global__ __launch_bounds__(256)
void bias_relu_kernel(float* __restrict__ vw, const float* __restrict__ b) {
  int i = blockIdx.x * 256 + threadIdx.x;
  if (i < NN * OUTD) vw[i] = fmaxf(vw[i] + b[i & (OUTD - 1)], 0.f);
}

// ---------------- fusion: per-view Q.K logit accumulation ----------------
__global__ __launch_bounds__(256)
void logits_kernel(const float* __restrict__ views, const float* __restrict__ Wq,
                   const float* __restrict__ bq, const float* __restrict__ Wk,
                   const float* __restrict__ bk, float* __restrict__ scal) {
  int v = blockIdx.y;
  const float* vw = views + (size_t)v * NN * OUTD;
  int i = blockIdx.x * 256 + threadIdx.x;
  float partial = 0.f;
  if (i < NN * HH) {
    int n = i >> 3, h = i & 7;
    const float* rowp = vw + (size_t)n * OUTD;
    const float* wq = Wq + h * OUTD;
    const float* wk = Wk + h * OUTD;
    float q = bq[h], k = bk[h];
#pragma unroll 4
    for (int c = 0; c < OUTD; ++c) { float x = rowp[c]; q = fmaf(x, wq[c], q); k = fmaf(x, wk[c], k); }
    partial = q * k;
  }
  __shared__ float red[256];
  red[threadIdx.x] = partial;
  __syncthreads();
  for (int st = 128; st > 0; st >>= 1) {
    if (threadIdx.x < st) red[threadIdx.x] += red[threadIdx.x + st];
    __syncthreads();
  }
  if (threadIdx.x == 0) atomicAdd(&scal[v], red[0]);
}

__global__ void softmax_w_kernel(float* __restrict__ scal) {
  float inv = rsqrtf((float)(FF * NN));
  float l0 = scal[0] * inv, l1 = scal[1] * inv, l2 = scal[2] * inv;
  float m = fmaxf(l0, fmaxf(l1, l2));
  float e0 = __expf(l0 - m), e1 = __expf(l1 - m), e2 = __expf(l2 - m);
  float si = 1.f / (e0 + e1 + e2);
  scal[3] = e0 * si; scal[4] = e1 * si; scal[5] = e2 * si;
}

// ---------------- omega = sigmoid(views2 . Wm + bm) accumulation ----------------
__global__ __launch_bounds__(256)
void omega_kernel(const float* __restrict__ views, const float* __restrict__ Wm,
                  float* __restrict__ scal) {
  int v = blockIdx.y;
  const float* vw = views + (size_t)v * NN * OUTD;
  float g = ALPHA_C * scal[3 + v] + (1.0f - ALPHA_C);
  float partial = 0.f;
  for (size_t i = (size_t)blockIdx.x * 256 + threadIdx.x; i < (size_t)NN * OUTD;
       i += (size_t)gridDim.x * 256)
    partial = fmaf(g * vw[i], Wm[i], partial);
  __shared__ float red[256];
  red[threadIdx.x] = partial;
  __syncthreads();
  for (int st = 128; st > 0; st >>= 1) {
    if (threadIdx.x < st) red[threadIdx.x] += red[threadIdx.x + st];
    __syncthreads();
  }
  if (threadIdx.x == 0) atomicAdd(&scal[6 + v], red[0]);
}

__global__ void omega_fin_kernel(float* __restrict__ scal, const float* __restrict__ bm) {
  for (int v = 0; v < 3; ++v) scal[9 + v] = 1.f / (1.f + __expf(-(scal[6 + v] + bm[0])));
}

// ---------------- final blend: writes d_out = [mv | result(3 views)] ----------------
__global__ __launch_bounds__(256)
void final_kernel(const float* __restrict__ views, const float* __restrict__ scal,
                  float* __restrict__ out) {
  int i = blockIdx.x * 256 + threadIdx.x;
  if (i >= NN * OUTD) return;
  const size_t sz = (size_t)NN * OUTD;
  float g0 = ALPHA_C * scal[3] + (1.0f - ALPHA_C);
  float g1 = ALPHA_C * scal[4] + (1.0f - ALPHA_C);
  float g2 = ALPHA_C * scal[5] + (1.0f - ALPHA_C);
  float a0 = g0 * views[i];
  float a1 = g1 * views[sz + i];
  float a2 = g2 * views[2 * sz + i];
  float mv = scal[9] * a0 + scal[10] * a1 + scal[11] * a2;
  out[i] = mv;
  out[sz + i]     = BETA_C * a0 + (1.0f - BETA_C) * mv;
  out[2 * sz + i] = BETA_C * a1 + (1.0f - BETA_C) * mv;
  out[3 * sz + i] = BETA_C * a2 + (1.0f - BETA_C) * mv;
}

// ---------------- host launch ----------------
extern "C" void kernel_launch(void* const* d_in, const int* in_sizes, int n_in,
                              void* d_out, int out_size, void* d_ws, size_t ws_size,
                              hipStream_t stream) {
  (void)n_in; (void)out_size; (void)ws_size;
  const float* x = (const float*)d_in[0];
  const int* srcv[3] = {(const int*)d_in[1], (const int*)d_in[3], (const int*)d_in[5]};
  const int* dstv[3] = {(const int*)d_in[2], (const int*)d_in[4], (const int*)d_in[6]};
  const float *Wv[3], *alv[3], *arv[3], *bv[3];
  for (int v = 0; v < 3; ++v) {
    Wv[v]  = (const float*)d_in[7 + 4 * v];
    alv[v] = (const float*)d_in[8 + 4 * v];
    arv[v] = (const float*)d_in[9 + 4 * v];
    bv[v]  = (const float*)d_in[10 + 4 * v];
  }
  const float* Wq = (const float*)d_in[19];
  const float* bq = (const float*)d_in[20];
  const float* Wk = (const float*)d_in[21];
  const float* bk = (const float*)d_in[22];
  const float* Wm = (const float*)d_in[23];
  const float* bm = (const float*)d_in[24];
  const int E = in_sizes[1];

  char* base = (char*)d_ws;
  _Float16* xb   = (_Float16*)(base + OFF_XB);
  _Float16* wb   = (_Float16*)(base + OFF_WB);
  float*    views= (float*)(base + OFF_VIEWS);
  float*    feat = (float*)(base + OFF_FEAT);
  float*    el   = (float*)(base + OFF_EL);
  float*    er   = (float*)(base + OFF_ER);
  unsigned* menc = (unsigned*)(base + OFF_M);
  float*    s    = (float*)(base + OFF_S);
  float*    scal = (float*)(base + OFF_SCAL);

  hipMemsetAsync(views, 0, (size_t)3 * NN * OUTD * sizeof(float), stream);
  hipMemsetAsync(scal, 0, 12 * sizeof(float), stream);

  conv_x_kernel<<<(NN * DD + 255) / 256, 256, 0, stream>>>(x, xb);
  conv_w_kernel<<<(3 * OUTD * DD + 255) / 256, 256, 0, stream>>>(Wv[0], Wv[1], Wv[2], wb);

  for (int v = 0; v < 3; ++v) {
    hipMemsetAsync(menc, 0, (size_t)NN * HH * sizeof(unsigned), stream);
    hipMemsetAsync(s, 0, (size_t)NN * HH * sizeof(float), stream);
    gat_feat_wmma<<<dim3(NN / 16 / 5), dim3(32, 5), 0, stream>>>(xb, wb + (size_t)v * OUTD * DD, feat);
    elr_kernel<<<(NN * HH + 255) / 256, 256, 0, stream>>>(feat, alv[v], arv[v], el, er);
    edge_max_kernel<<<(E * HH + 255) / 256, 256, 0, stream>>>(srcv[v], dstv[v], el, er, menc, E);
    edge_sum_kernel<<<(E * HH + 255) / 256, 256, 0, stream>>>(srcv[v], dstv[v], el, er, menc, s, E);
    edge_agg_kernel<<<(E + 7) / 8, 256, 0, stream>>>(srcv[v], dstv[v], el, er, menc, s, feat,
                                                     views + (size_t)v * NN * OUTD, E);
    bias_relu_kernel<<<(NN * OUTD + 255) / 256, 256, 0, stream>>>(views + (size_t)v * NN * OUTD, bv[v]);
  }

  logits_kernel<<<dim3((NN * HH + 255) / 256, 3), 256, 0, stream>>>(views, Wq, bq, Wk, bk, scal);
  softmax_w_kernel<<<1, 1, 0, stream>>>(scal);
  omega_kernel<<<dim3(1024, 3), 256, 0, stream>>>(views, Wm, scal);
  omega_fin_kernel<<<1, 1, 0, stream>>>(scal, bm);
  final_kernel<<<(NN * OUTD + 255) / 256, 256, 0, stream>>>(views, scal, (float*)d_out);
}
